// MegNet_layer_69947837383491
// MI455X (gfx1250) — matrix operand order repaired and checked by reading, not compiled, not run on hardware
//
#include <hip/hip_runtime.h>

// ---------------- problem constants ----------------
#define NB    100        // graphs
#define NPG   500        // nodes / graph
#define EPG   8000       // edges / graph
#define NN    (NB*NPG)   // 50000 nodes
#define NE    (NB*EPG)   // 800000 edges
#define DD    32         // in dim
#define HH    64         // hidden dim

typedef __attribute__((ext_vector_type(16))) _Float16 v16h;
typedef __attribute__((ext_vector_type(8)))  float    v8f;

// leaky relu (slope 0.01) as a 2-op max form: lrelu(x) = max(x, 0.01x)
__device__ __forceinline__ float lrelu(float x) { return fmaxf(x, 0.01f * x); }

__device__ __forceinline__ v8f wmma16(v16h a, v16h b, v8f c) {
  // D = A(16x32 f16) * B(32x16 f16) + C(16x16 f32)
  return __builtin_amdgcn_wmma_f32_16x16x32_f16(false, a, false, b, (short)0, c, false, false);
}

// ---- A fragment (16x32 tile, M = lane&15) from row-major f32 [*, ldk] ----
// ISA 16-bit A layout: lanes<16 halves 0..7 -> K 0..7, halves 8..15 -> K 16..23
//                      lanes>=16: +8 on both runs.
__device__ __forceinline__ v16h a_frag_f32(const float* X, int ldk, int row, int kbase, int lane) {
  int koff = kbase + ((lane & 16) ? 8 : 0);
  const float* p = X + (size_t)row * ldk + koff;
  v16h a;
#pragma unroll
  for (int h = 0; h < 8; ++h) { a[h] = (_Float16)p[h]; a[h + 8] = (_Float16)p[16 + h]; }
  return a;
}

// same, from f16 source: two contiguous 16-byte runs per lane
__device__ __forceinline__ v16h a_frag_f16(const _Float16* X, int ldk, int row, int kbase, int lane) {
  int koff = kbase + ((lane & 16) ? 8 : 0);
  const _Float16* p = X + (size_t)row * ldk + koff;
  v16h a;
#pragma unroll
  for (int h = 0; h < 8; ++h) { a[h] = p[h]; a[h + 8] = p[16 + h]; }
  return a;
}

template <int LD>
__device__ __forceinline__ v16h a_frag_lds(const _Float16 (&st)[16][LD], int row, int kbase, int lane) {
  int koff = kbase + ((lane & 16) ? 8 : 0);
  v16h a;
#pragma unroll
  for (int h = 0; h < 8; ++h) { a[h] = st[row][koff + h]; a[h + 8] = st[row][koff + 16 + h]; }
  return a;
}

// ---- B fragment from pre-transposed f16 weight Wt[N][K] (ldk = K) ----
// halves 0..7 -> K = kbase + h, halves 8..15 -> K = kbase + 16 + h (+8 for lanes>=16), N = lane&15
__device__ __forceinline__ v16h b_frag_f16t(const _Float16* __restrict__ Wt, int ldk,
                                            int kbase, int nbase, int lane) {
  int n  = nbase + (lane & 15);
  int k0 = kbase + ((lane & 16) ? 8 : 0);
  const _Float16* p = Wt + (size_t)n * ldk + k0;
  v16h b;
#pragma unroll
  for (int h = 0; h < 8; ++h) { b[h] = p[h]; b[h + 8] = p[16 + h]; }
  return b;
}

// ---- C (16x16 f32) -> LDS f16 stage, with bias + leaky relu ----
// C layout: VGPR r -> M = r (+8 for lanes>=16), N = lane&15.
template <int LD>
__device__ __forceinline__ void c_to_stage(_Float16 (&st)[16][LD], const v8f& c,
                                           const float* bias, int nbase, int lane) {
  int moff = (lane & 16) ? 8 : 0;
  int n = nbase + (lane & 15);
  float bv = bias[n];
#pragma unroll
  for (int r = 0; r < 8; ++r) st[r + moff][n] = (_Float16)lrelu(c[r] + bv);
}

// =====================================================================
// zero-fill
__global__ void megnet_zero_kernel(float* p, int n) {
  int i = blockIdx.x * blockDim.x + threadIdx.x;
  if (i < n) p[i] = 0.0f;
}

// =====================================================================
// weight prep: W (K x Nm, f32 row-major) -> Wt (Nm x K, f16)
__global__ void megnet_wprep_kernel(const float* __restrict__ W, _Float16* __restrict__ Wt,
                                    int K, int Nm) {
  int i = blockIdx.x * blockDim.x + threadIdx.x;
  if (i < K * Nm) {
    int k = i / Nm, n = i % Nm;
    Wt[(size_t)n * K + k] = (_Float16)W[i];
  }
}

// =====================================================================
// node-side MLP:  Y(16x64,f16) = lrelu(lrelu(X(16x32) W1(32x32) + b1) W2(32x64) + b2)
__global__ void megnet_node_mlp_kernel(const float* __restrict__ X,
                                       const _Float16* __restrict__ w1t, const float* __restrict__ b1,
                                       const _Float16* __restrict__ w2t, const float* __restrict__ b2,
                                       _Float16* __restrict__ Y) {
  int lane = threadIdx.x;
  int rowbase = blockIdx.x * 16;
  int row = lane & 15;
  __shared__ _Float16 s1[16][DD];

  v16h a0 = a_frag_f32(X, DD, rowbase + row, 0, lane);
  v8f c0 = {}, c1 = {};
  c0 = wmma16(a0, b_frag_f16t(w1t, DD, 0, 0, lane), c0);
  c1 = wmma16(a0, b_frag_f16t(w1t, DD, 0, 16, lane), c1);
  c_to_stage(s1, c0, b1, 0, lane);
  c_to_stage(s1, c1, b1, 16, lane);

  v16h a1 = a_frag_lds(s1, row, 0, lane);
  int moff = (lane & 16) ? 8 : 0;
#pragma unroll
  for (int nt = 0; nt < 4; ++nt) {
    v8f cc = {};
    cc = wmma16(a1, b_frag_f16t(w2t, DD, 0, nt * 16, lane), cc);
    int n = nt * 16 + (lane & 15);
    float bv = b2[n];
#pragma unroll
    for (int r = 0; r < 8; ++r)
      Y[(size_t)(rowbase + r + moff) * HH + n] = (_Float16)lrelu(cc[r] + bv);
  }
}

// =====================================================================
// fused edge kernel: e = ffe(edge_feats); e_new = eu([v[src]|v[dst]|e|u[dst]])
// writes out_e = e_new + edge_feats, accumulates ef_raw[dst] += e_new, deg[dst] += 1
__global__ void megnet_edge_kernel(const float* __restrict__ edge_feats,
                                   const int* __restrict__ src, const int* __restrict__ dst,
                                   const _Float16* __restrict__ fw1t, const float* __restrict__ fb1,
                                   const _Float16* __restrict__ fw2t, const float* __restrict__ fb2,
                                   const _Float16* __restrict__ ew1t, const float* __restrict__ eb1,
                                   const _Float16* __restrict__ ew2t, const float* __restrict__ eb2,
                                   const _Float16* __restrict__ v, const _Float16* __restrict__ u,
                                   float* __restrict__ out_e, float* __restrict__ ef_raw,
                                   float* __restrict__ deg) {
  int lane = threadIdx.x;
  int rowbase = blockIdx.x * 16;
  int row = lane & 15;
  int eidx = rowbase + row;
  __shared__ _Float16 s1[16][DD];      // ffe layer-1 output
  __shared__ _Float16 es[16][HH];      // e hidden (16x64)
  __shared__ _Float16 h1[16][HH];      // eu layer-1 output

  // ---- ffe layer 1 (16x32 @ 32x32) ----
  v16h a0 = a_frag_f32(edge_feats, DD, eidx, 0, lane);
  v8f c0 = {}, c1 = {};
  c0 = wmma16(a0, b_frag_f16t(fw1t, DD, 0, 0, lane), c0);
  c1 = wmma16(a0, b_frag_f16t(fw1t, DD, 0, 16, lane), c1);
  c_to_stage(s1, c0, fb1, 0, lane);
  c_to_stage(s1, c1, fb1, 16, lane);

  // ---- ffe layer 2 (16x32 @ 32x64) -> es ----
  v16h a1 = a_frag_lds(s1, row, 0, lane);
#pragma unroll
  for (int nt = 0; nt < 4; ++nt) {
    v8f cc = {};
    cc = wmma16(a1, b_frag_f16t(fw2t, DD, 0, nt * 16, lane), cc);
    c_to_stage(es, cc, fb2, nt * 16, lane);
  }

  // ---- eu layer 1: concat[v[s](64) | v[d](64) | e(64) | u[d](64)] (K=256) @ (256x64) ----
  int s = src[eidx];
  int d = dst[eidx];
  v8f acc[4] = {};
#pragma unroll
  for (int kb = 0; kb < 8; ++kb) {
    v16h af;
    if (kb < 2)      af = a_frag_f16(v, HH, s, kb * 32, lane);
    else if (kb < 4) af = a_frag_f16(v, HH, d, (kb - 2) * 32, lane);
    else if (kb < 6) af = a_frag_lds(es, row, (kb - 4) * 32, lane);
    else             af = a_frag_f16(u, HH, d, (kb - 6) * 32, lane);
#pragma unroll
    for (int nt = 0; nt < 4; ++nt)
      acc[nt] = wmma16(af, b_frag_f16t(ew1t, 4 * HH + 2 * DD + HH, kb * 32, nt * 16, lane), acc[nt]);
  }
#pragma unroll
  for (int nt = 0; nt < 4; ++nt) c_to_stage(h1, acc[nt], eb1, nt * 16, lane);

  // ---- eu layer 2 (16x64 @ 64x32) ----
  v8f o[2] = {};
#pragma unroll
  for (int kb = 0; kb < 2; ++kb) {
    v16h af = a_frag_lds(h1, row, kb * 32, lane);
#pragma unroll
    for (int nt = 0; nt < 2; ++nt)
      o[nt] = wmma16(af, b_frag_f16t(ew2t, HH, kb * 32, nt * 16, lane), o[nt]);
  }

  // ---- epilogue: residual + per-dst scatter-add ----
  int moff = (lane & 16) ? 8 : 0;
#pragma unroll
  for (int nt = 0; nt < 2; ++nt) {
    int n = nt * 16 + (lane & 15);
    float bv = eb2[n];
#pragma unroll
    for (int r = 0; r < 8; ++r) {
      int er = rowbase + r + moff;
      float en = lrelu(o[nt][r] + bv);
      out_e[(size_t)er * DD + n] = en + edge_feats[(size_t)er * DD + n];
      atomicAdd(&ef_raw[(size_t)dst[er] * DD + n], en);
    }
  }
  if (lane < 16) atomicAdd(&deg[d], 1.0f);
}

// =====================================================================
// per-graph: edge_pool = sum(ef_raw)/EPG; graph_pool = sum(u)/NPG; ef /= max(deg,1)
__global__ void megnet_graph_reduce_kernel(float* __restrict__ ef,          // in: raw sums, out: means
                                           const float* __restrict__ deg,
                                           const _Float16* __restrict__ u,
                                           float* __restrict__ edge_pool,
                                           float* __restrict__ graph_pool) {
  int g = blockIdx.x;
  int t = threadIdx.x;  // 128 threads
  float acc = 0.0f;
  if (t < DD) {
    for (int i = 0; i < NPG; ++i) acc += ef[(size_t)(g * NPG + i) * DD + t];
    edge_pool[g * DD + t] = acc * (1.0f / (float)EPG);
  } else if (t < DD + HH) {
    int c = t - DD;
    for (int i = 0; i < NPG; ++i) acc += (float)u[(size_t)(g * NPG + i) * HH + c];
    graph_pool[g * HH + c] = acc * (1.0f / (float)NPG);
  }
  __syncthreads();
  for (int idx = t; idx < NPG * DD; idx += blockDim.x) {
    int node = g * NPG + idx / DD;
    int c = idx % DD;
    float dg = deg[node];
    ef[(size_t)node * DD + c] = ef[(size_t)node * DD + c] / fmaxf(dg, 1.0f);
  }
}

// =====================================================================
// node update: n_new = nu([v(64)|ef(32)|u(64)]) ; out_n = n_new + node_feats;
// node_pool_raw[g] += n_new
__global__ void megnet_node_update_kernel(const _Float16* __restrict__ v,
                                          const float* __restrict__ ef,
                                          const _Float16* __restrict__ u,
                                          const _Float16* __restrict__ w1t, const float* __restrict__ b1,
                                          const _Float16* __restrict__ w2t, const float* __restrict__ b2,
                                          const float* __restrict__ node_feats,
                                          float* __restrict__ out_n,
                                          float* __restrict__ node_pool_raw) {
  int lane = threadIdx.x;
  int rowbase = blockIdx.x * 16;
  int row = lane & 15;
  int node = rowbase + row;
  __shared__ _Float16 h1[16][HH];

  v8f acc[4] = {};
#pragma unroll
  for (int kb = 0; kb < 5; ++kb) {
    v16h af;
    if (kb < 2)       af = a_frag_f16(v, HH, node, kb * 32, lane);
    else if (kb == 2) af = a_frag_f32(ef, DD, node, 0, lane);
    else              af = a_frag_f16(u, HH, node, (kb - 3) * 32, lane);
#pragma unroll
    for (int nt = 0; nt < 4; ++nt)
      acc[nt] = wmma16(af, b_frag_f16t(w1t, 2 * HH + DD, kb * 32, nt * 16, lane), acc[nt]);
  }
#pragma unroll
  for (int nt = 0; nt < 4; ++nt) c_to_stage(h1, acc[nt], b1, nt * 16, lane);

  v8f o[2] = {};
#pragma unroll
  for (int kb = 0; kb < 2; ++kb) {
    v16h af = a_frag_lds(h1, row, kb * 32, lane);
#pragma unroll
    for (int nt = 0; nt < 2; ++nt)
      o[nt] = wmma16(af, b_frag_f16t(w2t, HH, kb * 32, nt * 16, lane), o[nt]);
  }

  int moff = (lane & 16) ? 8 : 0;
#pragma unroll
  for (int nt = 0; nt < 2; ++nt) {
    int n = nt * 16 + (lane & 15);
    float bv = b2[n];
#pragma unroll
    for (int r = 0; r < 8; ++r) {
      int nr = rowbase + r + moff;
      float nv = lrelu(o[nt][r] + bv);
      out_n[(size_t)nr * DD + n] = nv + node_feats[(size_t)nr * DD + n];
      atomicAdd(&node_pool_raw[(nr / NPG) * DD + n], nv);
    }
  }
}

// =====================================================================
// graph MLP (B=100 rows, ~1 MFLOP — scalar is fine):
// g_new = mlp2([node_pool_raw/NPG | edge_pool | graph_pool], gu)
__global__ void megnet_graph_mlp_kernel(const float* __restrict__ node_pool_raw,
                                        const float* __restrict__ edge_pool,
                                        const float* __restrict__ graph_pool,
                                        const float* __restrict__ w1, const float* __restrict__ b1,
                                        const float* __restrict__ w2, const float* __restrict__ b2,
                                        float* __restrict__ g_new) {
  int g = blockIdx.x;
  int t = threadIdx.x;  // 64 threads
  __shared__ float xin[2 * DD + HH];
  __shared__ float hid[HH];
  for (int i = t; i < 2 * DD + HH; i += 64) {
    float val;
    if (i < DD)          val = node_pool_raw[g * DD + i] * (1.0f / (float)NPG);
    else if (i < 2 * DD) val = edge_pool[g * DD + (i - DD)];
    else                 val = graph_pool[g * HH + (i - 2 * DD)];
    xin[i] = val;
  }
  __syncthreads();
  float a = b1[t];
  for (int k = 0; k < 2 * DD + HH; ++k) a += xin[k] * w1[k * HH + t];
  hid[t] = lrelu(a);
  __syncthreads();
  if (t < DD) {
    float o = b2[t];
    for (int k = 0; k < HH; ++k) o += hid[k] * w2[k * DD + t];
    g_new[g * DD + t] = lrelu(o);
  }
}

// =====================================================================
// u_new broadcast + residual
__global__ void megnet_bcast_u_kernel(const float* __restrict__ g_new,
                                      const float* __restrict__ graph_feats,
                                      float* __restrict__ out_u) {
  int i = blockIdx.x * blockDim.x + threadIdx.x;
  if (i < NN * DD) {
    int node = i >> 5;           // /32
    int g = node / NPG;
    out_u[i] = g_new[g * DD + (i & 31)] + graph_feats[i];
  }
}

// =====================================================================
static inline void* ws_alloc(char* ws, size_t& off, size_t bytes) {
  void* p = ws + off;
  off += (bytes + 255) & ~(size_t)255;
  return p;
}

extern "C" void kernel_launch(void* const* d_in, const int* in_sizes, int n_in,
                              void* d_out, int out_size, void* d_ws, size_t ws_size,
                              hipStream_t stream) {
  (void)in_sizes; (void)n_in; (void)out_size; (void)ws_size;

  const float* node_feats  = (const float*)d_in[0];
  const float* edge_feats  = (const float*)d_in[1];
  const float* graph_feats = (const float*)d_in[2];
  const int*   src         = (const int*)d_in[3];
  const int*   dst         = (const int*)d_in[4];
  // d_in[5] node_graph, d_in[6] edge_graph: implicit (contiguous blocks)
  const float* ffn_w1 = (const float*)d_in[7];  const float* ffn_b1 = (const float*)d_in[8];
  const float* ffn_w2 = (const float*)d_in[9];  const float* ffn_b2 = (const float*)d_in[10];
  const float* ffe_w1 = (const float*)d_in[11]; const float* ffe_b1 = (const float*)d_in[12];
  const float* ffe_w2 = (const float*)d_in[13]; const float* ffe_b2 = (const float*)d_in[14];
  const float* ffg_w1 = (const float*)d_in[15]; const float* ffg_b1 = (const float*)d_in[16];
  const float* ffg_w2 = (const float*)d_in[17]; const float* ffg_b2 = (const float*)d_in[18];
  const float* eu_w1  = (const float*)d_in[19]; const float* eu_b1  = (const float*)d_in[20];
  const float* eu_w2  = (const float*)d_in[21]; const float* eu_b2  = (const float*)d_in[22];
  const float* nu_w1  = (const float*)d_in[23]; const float* nu_b1  = (const float*)d_in[24];
  const float* nu_w2  = (const float*)d_in[25]; const float* nu_b2  = (const float*)d_in[26];
  const float* gu_w1  = (const float*)d_in[27]; const float* gu_b1  = (const float*)d_in[28];
  const float* gu_w2  = (const float*)d_in[29]; const float* gu_b2  = (const float*)d_in[30];

  // ---- workspace layout (~19.6 MB) ----
  char* ws = (char*)d_ws;
  size_t off = 0;
  _Float16* v_h16 = (_Float16*)ws_alloc(ws, off, (size_t)NN * HH * 2);
  _Float16* u_h16 = (_Float16*)ws_alloc(ws, off, (size_t)NN * HH * 2);
  float* ef_raw   = (float*)ws_alloc(ws, off, (size_t)(NN * DD + NN + NB * DD) * 4); // ef | deg | npool contiguous
  float* deg      = ef_raw + (size_t)NN * DD;
  float* npool    = deg + NN;
  float* epool    = (float*)ws_alloc(ws, off, (size_t)NB * DD * 4);
  float* gpool    = (float*)ws_alloc(ws, off, (size_t)NB * HH * 4);
  float* g_new    = (float*)ws_alloc(ws, off, (size_t)NB * DD * 4);
  // f16 transposed weights (~94 KB total; stay hot in cache across all tiles)
  _Float16* ffn_w1t = (_Float16*)ws_alloc(ws, off, (size_t)DD * DD * 2);
  _Float16* ffn_w2t = (_Float16*)ws_alloc(ws, off, (size_t)DD * HH * 2);
  _Float16* ffe_w1t = (_Float16*)ws_alloc(ws, off, (size_t)DD * DD * 2);
  _Float16* ffe_w2t = (_Float16*)ws_alloc(ws, off, (size_t)DD * HH * 2);
  _Float16* ffg_w1t = (_Float16*)ws_alloc(ws, off, (size_t)DD * DD * 2);
  _Float16* ffg_w2t = (_Float16*)ws_alloc(ws, off, (size_t)DD * HH * 2);
  _Float16* eu_w1t  = (_Float16*)ws_alloc(ws, off, (size_t)(4 * HH + 2 * DD + HH) * HH * 2); // 384x64
  _Float16* eu_w2t  = (_Float16*)ws_alloc(ws, off, (size_t)HH * DD * 2);
  _Float16* nu_w1t  = (_Float16*)ws_alloc(ws, off, (size_t)(2 * HH + DD) * HH * 2);          // 160x64
  _Float16* nu_w2t  = (_Float16*)ws_alloc(ws, off, (size_t)HH * DD * 2);

  float* out_n = (float*)d_out;                       // N*32
  float* out_e = out_n + (size_t)NN * DD;             // E*32
  float* out_u = out_e + (size_t)NE * DD;             // N*32

  // 0) weight prep: f32 [K,N] -> f16 [N,K]
  struct { const float* W; _Float16* Wt; int K, Nm; } wp[10] = {
      {ffn_w1, ffn_w1t, DD, DD}, {ffn_w2, ffn_w2t, DD, HH},
      {ffe_w1, ffe_w1t, DD, DD}, {ffe_w2, ffe_w2t, DD, HH},
      {ffg_w1, ffg_w1t, DD, DD}, {ffg_w2, ffg_w2t, DD, HH},
      {eu_w1,  eu_w1t,  4 * HH + 2 * DD + HH, HH},  // 384x64
      {eu_w2,  eu_w2t,  HH, DD},
      {nu_w1,  nu_w1t,  2 * HH + DD, HH},            // 160x64
      {nu_w2,  nu_w2t,  HH, DD}};
  for (int i = 0; i < 10; ++i) {
    int cnt = wp[i].K * wp[i].Nm;
    megnet_wprep_kernel<<<(cnt + 255) / 256, 256, 0, stream>>>(wp[i].W, wp[i].Wt, wp[i].K, wp[i].Nm);
  }

  // 1) zero accumulators (ef_raw, deg, npool are contiguous)
  int zcount = NN * DD + NN + NB * DD;
  megnet_zero_kernel<<<(zcount + 255) / 256, 256, 0, stream>>>(ef_raw, zcount);

  // 2) v = ffn(node_feats), u = ffg(graph_feats)  — 1 wave / 16-row tile
  megnet_node_mlp_kernel<<<NN / 16, 32, 0, stream>>>(node_feats, ffn_w1t, ffn_b1, ffn_w2t, ffn_b2, v_h16);
  megnet_node_mlp_kernel<<<NN / 16, 32, 0, stream>>>(graph_feats, ffg_w1t, ffg_b1, ffg_w2t, ffg_b2, u_h16);

  // 3) fused ffe + edge update (50k tiles)
  megnet_edge_kernel<<<NE / 16, 32, 0, stream>>>(edge_feats, src, dst,
      ffe_w1t, ffe_b1, ffe_w2t, ffe_b2, eu_w1t, eu_b1, eu_w2t, eu_b2,
      v_h16, u_h16, out_e, ef_raw, deg);

  // 4) per-graph pools + ef mean
  megnet_graph_reduce_kernel<<<NB, 128, 0, stream>>>(ef_raw, deg, u_h16, epool, gpool);

  // 5) node update
  megnet_node_update_kernel<<<NN / 16, 32, 0, stream>>>(v_h16, ef_raw, u_h16,
      nu_w1t, nu_b1, nu_w2t, nu_b2, node_feats, out_n, npool);

  // 6) graph MLP
  megnet_graph_mlp_kernel<<<NB, 64, 0, stream>>>(npool, epool, gpool,
      gu_w1, gu_b1, gu_w2, gu_b2, g_new);

  // 7) broadcast u_new + residual
  megnet_bcast_u_kernel<<<(NN * DD + 255) / 256, 256, 0, stream>>>(g_new, graph_feats, out_u);
}
